// SuperNodePool_20478404067768
// MI455X (gfx1250) — matrix operand order repaired and checked by reading, not compiled, not run on hardware
//
#include <hip/hip_runtime.h>
#include <hip/hip_bf16.h>
#include <math.h>

#define HID 128
#define HEADS 8
#define DH 16
#define GSEG 4096
#define LN_EPS 1e-5f

typedef __attribute__((ext_vector_type(16))) __bf16    v16bf;
typedef __attribute__((ext_vector_type(8)))  float     v8f;
typedef __attribute__((ext_vector_type(4)))  float     v4f;
typedef __attribute__((ext_vector_type(8)))  unsigned  v8u;

// workspace layout (dwords)
#define OFF_BV 0      // 8192 u32 : Wv^T packed as bf16 B-fragments [nt=8][kf=4][lane=32][8 dw]
#define OFF_BS 8192   // 1024 u32 : score matrix A (128x16, cols 8..15 = 0) as B-fragments
#define OFF_CB 9216   // 8 f32   : per-head score bias c_h = q_h . bk_h
#define OFF_GS 9232   // 4097 i32: segment start offsets (lower bounds of batch)
// total ~53 KB of d_ws

__device__ __forceinline__ unsigned f2bf(float f) {
  unsigned u = __float_as_uint(f);
  return (u + 0x7FFFu + ((u >> 16) & 1u)) >> 16;   // round-to-nearest-even
}
__device__ __forceinline__ unsigned pk2(float lo, float hi) {
  return f2bf(lo) | (f2bf(hi) << 16);
}
__device__ __forceinline__ float bperm(int idx, float v) {
  return __int_as_float(__builtin_amdgcn_ds_bpermute(idx, __float_as_int(v)));
}
__device__ __forceinline__ v4f ldnt4(const float* p) {
  return __builtin_nontemporal_load((const v4f*)p);
}

// ---------------------------------------------------------------------------
// Prep 1: fold query into Wk, pack score-matrix and Wv into WMMA B-fragment
// layout (bf16) in workspace. One block of 128 threads.
// B-fragment layout (16x16x32 bf16, B is 32K x 16N): lane L holds N = L&15,
// K base = (L>=16)?8:0, dword j = K pair base+2j (j<4) or 16+base+2(j-4) (j>=4).
// ---------------------------------------------------------------------------
__global__ void sp_prep(const float* __restrict__ query,
                        const float* __restrict__ ipw,
                        const float* __restrict__ ipb,
                        unsigned* __restrict__ ws) {
  __shared__ float q[HID];
  __shared__ float a[HEADS][HID];
  const int tid = threadIdx.x;

  // q = (query @ Wq^T + bq) / sqrt(DH)
  float acc = ipb[tid];
  for (int c = 0; c < HID; ++c) acc += query[c] * ipw[tid * HID + c];
  q[tid] = acc * 0.25f;
  __syncthreads();

  // a[h][c] = sum_d q[h*16+d] * Wk[h*16+d][c]
  for (int h = 0; h < HEADS; ++h) {
    float s = 0.f;
    for (int dd = 0; dd < DH; ++dd)
      s += q[h * DH + dd] * ipw[(HID + h * DH + dd) * HID + tid];
    a[h][tid] = s;
  }
  __syncthreads();

  if (tid < HEADS) {
    float s = 0.f;
    for (int dd = 0; dd < DH; ++dd) s += q[tid * DH + dd] * ipb[HID + tid * DH + dd];
    ((float*)ws)[OFF_CB + tid] = s;
  }

  // score B-fragments: B[k][n] = a_n[k] for n<8 else 0
  for (int i = tid; i < 1024; i += 128) {
    int kf = i >> 8, lane = (i >> 3) & 31, j = i & 7;
    int nn = lane & 15;
    int k0 = kf * 32 + ((j >> 2) << 4) + ((lane >> 4) << 3) + 2 * (j & 3);
    float lo = (nn < 8) ? a[nn][k0] : 0.f;
    float hi = (nn < 8) ? a[nn][k0 + 1] : 0.f;
    ws[OFF_BS + i] = pk2(lo, hi);
  }
  // Wv B-fragments: B_nt[k][n] = Wv[nt*16+n][k]
  for (int i = tid; i < 8192; i += 128) {
    int nt = i >> 10, kf = (i >> 8) & 3, lane = (i >> 3) & 31, j = i & 7;
    int nn = nt * 16 + (lane & 15);
    int k0 = kf * 32 + ((j >> 2) << 4) + ((lane >> 4) << 3) + 2 * (j & 3);
    const float* wr = ipw + (size_t)(2 * HID + nn) * HID;
    ws[OFF_BV + i] = pk2(wr[k0], wr[k0 + 1]);
  }
}

// ---------------------------------------------------------------------------
// Prep 2: segment boundaries — lower_bound of g in the sorted batch array.
// ---------------------------------------------------------------------------
__global__ void sp_bounds(const int* __restrict__ batch, int* __restrict__ gs, int n) {
  int g = blockIdx.x * blockDim.x + threadIdx.x;
  if (g > GSEG) return;
  int lo = 0, hc = n;
  while (lo < hc) {
    int mid = (lo + hc) >> 1;
    if (batch[mid] < g) lo = mid + 1; else hc = mid;
  }
  gs[g] = lo;
}

// ---------------------------------------------------------------------------
// Main: one workgroup (4 waves) per segment. Online softmax + WMMA V-GEMM.
// All B fragments (Wv + score matrix) live in VGPRs for the whole kernel,
// so the 36 WMMAs per 16-row tile have no in-loop memory dependency on B.
// ---------------------------------------------------------------------------
__global__ __launch_bounds__(128, 1) void sp_main(
    const float* __restrict__ x, const float* __restrict__ ipb,
    const float* __restrict__ opw, const float* __restrict__ opb,
    const float* __restrict__ lnw, const float* __restrict__ lnb,
    const unsigned* __restrict__ ws, const int* __restrict__ gs,
    float* __restrict__ out, int n) {
  __shared__ __align__(32) float wbufT[4][8][16];  // [wave][head][row] weights
  __shared__ float wacc[4][8][16];
  __shared__ float wdd[4][8];
  __shared__ float wmm[4][8];
  __shared__ float pooled[HID];
  __shared__ float attn[HID];

  const int tid = threadIdx.x;
  const int lane = tid & 31;
  const int wv = tid >> 5;
  const int mrow = lane & 15;   // M-row (A) / N-col (B,C) owned by this lane
  const int hi = lane >> 4;
  const int g = blockIdx.x;

  // loop-invariant B fragments -> registers (L2-resident, shared by all blocks)
  v16bf Bv[8][4];
#pragma unroll
  for (int nt = 0; nt < 8; ++nt)
#pragma unroll
    for (int kf = 0; kf < 4; ++kf)
      Bv[nt][kf] = __builtin_bit_cast(
          v16bf, *(const v8u*)(ws + OFF_BV + ((nt * 4 + kf) * 32 + lane) * 8));
  v16bf Bsf[4];
#pragma unroll
  for (int kf = 0; kf < 4; ++kf)
    Bsf[kf] = __builtin_bit_cast(v16bf, *(const v8u*)(ws + OFF_BS + (kf * 32 + lane) * 8));

  const float cb = ((const float*)ws)[OFF_CB + (mrow & 7)];
  float bvr[8];
#pragma unroll
  for (int nt = 0; nt < 8; ++nt) bvr[nt] = ipb[2 * HID + nt * 16 + mrow];

  float m[8], dsum[8], acc[8];
#pragma unroll
  for (int hh = 0; hh < 8; ++hh) { m[hh] = -3.0e38f; dsum[hh] = 0.f; acc[hh] = 0.f; }

  const int s = gs[g], e = gs[g + 1];
  const int cnt = e - s;
  const int idxswap = (lane ^ 16) << 2;

  for (int t = wv; t * 16 < cnt; t += 4) {
    const int base = s + t * 16;
    const int row = base + mrow;
    const float* xr = x + (size_t)(row < n ? row : n - 1) * HID;
    {
      const int nrow = base + 64 + mrow;            // next tile for this wave
      if (nrow < e) __builtin_prefetch(x + (size_t)nrow * HID, 0, 0);
    }

    // X tile in A-fragment layout (bf16): lane holds row mrow,
    // cols kf*32 + hi*8 + {0..7} and +16 + {0..7}. Non-temporal (streamed).
    v16bf A[4];
#pragma unroll
    for (int kf = 0; kf < 4; ++kf) {
      const int c0 = kf * 32 + hi * 8;
      v4f f0 = ldnt4(xr + c0);
      v4f f1 = ldnt4(xr + c0 + 4);
      v4f g0 = ldnt4(xr + c0 + 16);
      v4f g1 = ldnt4(xr + c0 + 20);
      v8u au;
      au[0] = pk2(f0[0], f0[1]); au[1] = pk2(f0[2], f0[3]);
      au[2] = pk2(f1[0], f1[1]); au[3] = pk2(f1[2], f1[3]);
      au[4] = pk2(g0[0], g0[1]); au[5] = pk2(g0[2], g0[3]);
      au[6] = pk2(g1[0], g1[1]); au[7] = pk2(g1[2], g1[3]);
      A[kf] = __builtin_bit_cast(v16bf, au);
    }

    // scores tile: D[M][h] = X @ A_score ; C-layout lane holds head = mrow
    v8f sc = {};
#pragma unroll
    for (int kf = 0; kf < 4; ++kf)
      sc = __builtin_amdgcn_wmma_f32_16x16x32_bf16(false, A[kf], false, Bsf[kf],
                                                   (short)0, sc, false, false);
#pragma unroll
    for (int j = 0; j < 8; ++j) {
      const int rj = base + j + hi * 8;
      sc[j] = (rj < e) ? (sc[j] + cb) : -3.0e38f;
    }

    // per-head tile max -> broadcast to all lanes
    float tmax = sc[0];
#pragma unroll
    for (int j = 1; j < 8; ++j) tmax = fmaxf(tmax, sc[j]);
    tmax = fmaxf(tmax, bperm(idxswap, tmax));       // merge row halves
#pragma unroll
    for (int hh = 0; hh < 8; ++hh) {
      const float tm = bperm(hh << 2, tmax);        // head hh lives in lane hh
      const float mn = fmaxf(m[hh], tm);
      const float f = __expf(m[hh] - mn);
      m[hh] = mn; dsum[hh] *= f; acc[hh] *= f;
    }

    // weights -> LDS, transposed: wbufT[wave][head][row]; one v8f store
    if (mrow < 8) {
      const float mh = m[mrow];
      v8f w8;
#pragma unroll
      for (int j = 0; j < 8; ++j) w8[j] = __expf(sc[j] - mh);
      *(v8f*)&wbufT[wv][mrow][hi * 8] = w8;
    }
    __builtin_amdgcn_wave_barrier();                // keep DS store->load order

    // V tiles: one 16x16 tile per head; weighted row-reduce into acc
#pragma unroll
    for (int nt = 0; nt < 8; ++nt) {
      v8f vc;
#pragma unroll
      for (int j = 0; j < 8; ++j) vc[j] = bvr[nt];
#pragma unroll
      for (int kf = 0; kf < 4; ++kf)
        vc = __builtin_amdgcn_wmma_f32_16x16x32_bf16(false, A[kf], false, Bv[nt][kf],
                                                     (short)0, vc, false, false);
      const v8f wrn = *(const v8f*)&wbufT[wv][nt][hi * 8];
      float a2 = acc[nt], dsn = dsum[nt];
#pragma unroll
      for (int j = 0; j < 8; ++j) { a2 = fmaf(wrn[j], vc[j], a2); dsn += wrn[j]; }
      acc[nt] = a2;
      dsum[nt] = dsn;
    }
  }

  // combine the two row-halves within the wave
#pragma unroll
  for (int nt = 0; nt < 8; ++nt) {
    acc[nt] += bperm(idxswap, acc[nt]);
    dsum[nt] += bperm(idxswap, dsum[nt]);
  }
  if (lane < 16) {
#pragma unroll
    for (int nt = 0; nt < 8; ++nt) wacc[wv][nt][lane] = acc[nt];
    if (lane < 8) { wdd[wv][lane] = dsum[lane]; wmm[wv][lane] = m[lane]; }
  }
  __syncthreads();

  // flash-merge across the 4 waves; thread tid owns output column tid
  {
    const int nt2 = tid >> 4, nn2 = tid & 15;
    float Mx = wmm[0][nt2];
#pragma unroll
    for (int w = 1; w < 4; ++w) Mx = fmaxf(Mx, wmm[w][nt2]);
    float D = 0.f, Aa = 0.f;
#pragma unroll
    for (int w = 0; w < 4; ++w) {
      const float f = __expf(wmm[w][nt2] - Mx);
      D = fmaf(f, wdd[w][nt2], D);
      Aa = fmaf(f, wacc[w][nt2][nn2], Aa);
    }
    pooled[tid] = Aa / D;
  }
  __syncthreads();

  // out-projection + layernorm
  float att = opb[tid];
  for (int c = 0; c < HID; ++c) att = fmaf(pooled[c], opw[tid * HID + c], att);
  attn[tid] = att;
  __syncthreads();
  float mu = 0.f, s2 = 0.f;
  for (int c = 0; c < HID; ++c) { const float v = attn[c]; mu += v; s2 = fmaf(v, v, s2); }
  mu *= (1.0f / HID);
  const float var = s2 * (1.0f / HID) - mu * mu;
  out[g * HID + tid] = (att - mu) * rsqrtf(var + LN_EPS) * lnw[tid] + lnb[tid];
}

// ---------------------------------------------------------------------------
extern "C" void kernel_launch(void* const* d_in, const int* in_sizes, int n_in,
                              void* d_out, int out_size, void* d_ws, size_t ws_size,
                              hipStream_t stream) {
  const float* x     = (const float*)d_in[0];
  const int*   batch = (const int*)d_in[1];
  const float* query = (const float*)d_in[2];
  const float* ipw   = (const float*)d_in[3];
  const float* ipb   = (const float*)d_in[4];
  const float* opw   = (const float*)d_in[5];
  const float* opb   = (const float*)d_in[6];
  const float* lnw   = (const float*)d_in[7];
  const float* lnb   = (const float*)d_in[8];
  const int n = in_sizes[0] / HID;

  unsigned* ws = (unsigned*)d_ws;
  int* gsp = (int*)(ws + OFF_GS);

  sp_prep<<<1, 128, 0, stream>>>(query, ipw, ipb, ws);
  sp_bounds<<<(GSEG + 1 + 127) / 128, 128, 0, stream>>>(batch, gsp, n);
  sp_main<<<GSEG, 128, 0, stream>>>(x, ipb, opw, opb, lnw, lnb, ws, gsp,
                                    (float*)d_out, n);
}
